// CifarNet_1_60155311948284
// MI455X (gfx1250) — matrix operand order repaired and checked by reading, not compiled, not run on hardware
//
#include <hip/hip_runtime.h>
#include <hip/hip_bf16.h>

typedef __attribute__((ext_vector_type(16))) _Float16 v16h;
typedef __attribute__((ext_vector_type(8)))  _Float16 v8h;
typedef __attribute__((ext_vector_type(8)))  float    v8f;

#define NFEAT 256
#define DIN   64
#define DOUT  20
#define KW2INV 50.0f            // 1/(2*0.1^2)

__device__ __forceinline__ void wave_fence() {
  __builtin_amdgcn_wave_barrier();
  asm volatile("" ::: "memory");
}

// A fragment, 16x32 f16 (MxK). Source: row-major [m][k], stride ld halves.
// Lane l: M = mrow + (l&15); halves 0-7 <- K = k0 + (l>>4)*8 + 0..7,
//         halves 8-15 <- K = k0 + 16 + (l>>4)*8 + 0..7.
__device__ __forceinline__ v16h load_fragA(const _Float16* base, int ld,
                                           int mrow, int k0, int lane) {
  const int m = mrow + (lane & 15);
  const int g = lane >> 4;
  const _Float16* p = base + m * ld + k0 + g * 8;
  union { v16h v; v8h h[2]; } u;
  u.h[0] = *(const v8h*)(p);
  u.h[1] = *(const v8h*)(p + 16);
  return u.v;
}

// B fragment, 32x16 f16 (KxN), with B[k][n] = src[n][k] (src row-major by n).
// Lane l: N = ncol + (l&15); halves 0-7 <- K = k0 + (l>>4)*8 + 0..7,
//         halves 8-15 <- K = k0 + 16 + (l>>4)*8 + 0..7.
__device__ __forceinline__ v16h load_fragB(const _Float16* base, int ld,
                                           int ncol, int k0, int lane) {
  const int n = ncol + (lane & 15);
  const int g = lane >> 4;
  const _Float16* p = base + n * ld + k0 + g * 8;
  union { v16h v; v8h h[2]; } u;
  u.h[0] = *(const v8h*)(p);
  u.h[1] = *(const v8h*)(p + 16);
  return u.v;
}

// ---------------------------------------------------------------------------
// Kernel 1: per batch item -> Y = W^T K W (20x20) into workspace.
// ---------------------------------------------------------------------------
__global__ __launch_bounds__(256)
void cifarnet_y_kernel(const float* __restrict__ x, const float* __restrict__ W,
                       float* __restrict__ Yout) {
  __shared__ _Float16 xh[DIN][NFEAT + 8];   // mean-centered x, f16 (stride 264)
  __shared__ float    sq[DIN];              // row squared norms (f32)
  __shared__ _Float16 Kh[DIN][72];          // RBF kernel matrix, f16
  __shared__ _Float16 Wth[32][72];          // W^T zero-padded: Wth[m][k]=W[k][m]
  __shared__ _Float16 Tt[32][72];           // (K*W)^T: Tt[d][i] = T[i][d]

  const int b    = blockIdx.x;
  const int tid  = threadIdx.x;
  const int lane = tid & 31;
  const int wave = tid >> 5;

  // -------- Stage 1: mean-center, f32 -> f16 into LDS, row sq-sums --------
  {
    const int row = tid >> 2;               // 64 rows, 4 threads per row
    const int sub = tid & 3;                // 64 contiguous floats per thread
    const float* xr = x + ((size_t)b * DIN + row) * NFEAT + sub * 64;
    float4 v[16];
    float s = 0.0f;
#pragma unroll
    for (int i = 0; i < 16; ++i) {
      v[i] = ((const float4*)xr)[i];
      s += v[i].x + v[i].y + v[i].z + v[i].w;
    }
    s += __shfl_xor(s, 1);
    s += __shfl_xor(s, 2);
    const float mean = s * (1.0f / NFEAT);
    float q = 0.0f;
    _Float16* dst = &xh[row][sub * 64];
#pragma unroll
    for (int i = 0; i < 16; ++i) {
      const float a0 = v[i].x - mean, a1 = v[i].y - mean;
      const float a2 = v[i].z - mean, a3 = v[i].w - mean;
      q += a0 * a0 + a1 * a1 + a2 * a2 + a3 * a3;
      dst[4 * i + 0] = (_Float16)a0;  dst[4 * i + 1] = (_Float16)a1;
      dst[4 * i + 2] = (_Float16)a2;  dst[4 * i + 3] = (_Float16)a3;
    }
    q += __shfl_xor(q, 1);
    q += __shfl_xor(q, 2);
    if (sub == 0) sq[row] = q;
  }

  // -------- W^T into LDS, zero-padded to 32 rows --------
  for (int idx = tid; idx < 32 * 64; idx += 256) {
    const int m = idx >> 6, k = idx & 63;
    Wth[m][k] = (m < DOUT) ? (_Float16)W[k * DOUT + m] : (_Float16)0.0f;
  }
  __syncthreads();

  // -------- Stage 2: gram via WMMA, fused RBF epilogue -> Kh --------
  for (int t = wave; t < 16; t += 8) {
    const int mi = (t >> 2) * 16, ni = (t & 3) * 16;
    v8f acc = {};
#pragma unroll
    for (int k0 = 0; k0 < NFEAT; k0 += 32) {
      v16h a  = load_fragA(&xh[0][0], NFEAT + 8, mi, k0, lane);
      v16h bb = load_fragB(&xh[0][0], NFEAT + 8, ni, k0, lane);
      acc = __builtin_amdgcn_wmma_f32_16x16x32_f16(false, a, false, bb,
                                                   (short)0, acc, false, false);
    }
    const int nn    = ni + (lane & 15);
    const int mbase = mi + ((lane >> 4) << 3);
    const float sqn = sq[nn];
#pragma unroll
    for (int v = 0; v < 8; ++v) {
      const int mm = mbase + v;
      float d2 = (sq[mm] + sqn - 2.0f * acc[v]) * (1.0f / NFEAT);
      d2 = fmaxf(d2, 0.0f);
      Kh[mm][nn] = (_Float16)__expf(-KW2INV * d2);
    }
  }
  __syncthreads();

  // -------- Stage 3: T = K * W (64x64 @ 64x32), store transposed --------
  {
    const int mi = (wave >> 1) * 16, ni = (wave & 1) * 16;  // 8 tiles, 1/wave
    v8f acc = {};
#pragma unroll
    for (int k0 = 0; k0 < 64; k0 += 32) {
      v16h a  = load_fragA(&Kh[0][0],  72, mi, k0, lane);
      v16h bb = load_fragB(&Wth[0][0], 72, ni, k0, lane);
      acc = __builtin_amdgcn_wmma_f32_16x16x32_f16(false, a, false, bb,
                                                   (short)0, acc, false, false);
    }
    const int nn    = ni + (lane & 15);
    const int mbase = mi + ((lane >> 4) << 3);
#pragma unroll
    for (int v = 0; v < 8; ++v)
      Tt[nn][mbase + v] = (_Float16)acc[v];
  }
  __syncthreads();

  // -------- Stage 4: Y = W^T * T (32x64 @ 64x32), emit valid 20x20 --------
  if (wave < 4) {
    const int mi = (wave >> 1) * 16, ni = (wave & 1) * 16;
    v8f acc = {};
#pragma unroll
    for (int k0 = 0; k0 < 64; k0 += 32) {
      v16h a  = load_fragA(&Wth[0][0], 72, mi, k0, lane);
      v16h bb = load_fragB(&Tt[0][0],  72, ni, k0, lane);
      acc = __builtin_amdgcn_wmma_f32_16x16x32_f16(false, a, false, bb,
                                                   (short)0, acc, false, false);
    }
    const int dd    = ni + (lane & 15);
    const int cbase = mi + ((lane >> 4) << 3);
    if (dd < DOUT) {
      float* yb = Yout + (size_t)b * (DOUT * DOUT);
#pragma unroll
      for (int v = 0; v < 8; ++v) {
        const int cc = cbase + v;
        if (cc < DOUT) yb[cc * DOUT + dd] = acc[v];
      }
    }
  }
}

// ---------------------------------------------------------------------------
// Kernel 2: per wave32 -> Jacobi eigh of one 20x20 Y, matrix-log, linear head.
// ---------------------------------------------------------------------------
__global__ __launch_bounds__(256)
void cifarnet_eig_kernel(const float* __restrict__ Yin,
                         const float* __restrict__ lin_w,
                         const float* __restrict__ lin_b,
                         float* __restrict__ out) {
  __shared__ float A_s[8][DOUT][DOUT + 1];
  __shared__ float U_s[8][DOUT][DOUT + 1];
  __shared__ float l_s[8][DOUT];
  __shared__ float v_s[8][216];
  __shared__ float lw_s[10 * 210];
  __shared__ float lb_s[10];

  const int tid  = threadIdx.x;
  const int lane = tid & 31;
  const int w    = tid >> 5;
  const int b    = blockIdx.x * 8 + w;
  const int j    = lane;                    // column handled by this lane

  for (int i = tid; i < 2100; i += 256) lw_s[i] = lin_w[i];
  if (tid < 10) lb_s[tid] = lin_b[tid];

  const float* yb = Yin + (size_t)b * 400;
  for (int idx = lane; idx < 400; idx += 32) {
    const int i = idx / 20, jj = idx % 20;
    A_s[w][i][jj] = yb[idx];
    U_s[w][i][jj] = (i == jj) ? 1.0f : 0.0f;
  }
  __syncthreads();

  // Cyclic Jacobi: A <- J^T A J, U <- U J. All wave-uniform scalars.
  for (int sweep = 0; sweep < 10; ++sweep) {
    for (int p = 0; p < DOUT - 1; ++p) {
      for (int q = p + 1; q < DOUT; ++q) {
        wave_fence();
        const float app = A_s[w][p][p];
        const float aqq = A_s[w][q][q];
        const float apq = A_s[w][p][q];
        if (fabsf(apq) > 1e-12f) {
          const float th = 0.5f * (aqq - app) / apq;
          const float tt = ((th >= 0.0f) ? 1.0f : -1.0f) /
                           (fabsf(th) + sqrtf(th * th + 1.0f));
          const float c = rsqrtf(1.0f + tt * tt);
          const float s = tt * c;
          if (j < DOUT) {                    // row rotation (rows p,q)
            const float apj = A_s[w][p][j], aqj = A_s[w][q][j];
            A_s[w][p][j] = c * apj - s * aqj;
            A_s[w][q][j] = s * apj + c * aqj;
          }
          wave_fence();
          if (j < DOUT) {                    // col rotation (cols p,q) + U
            const float aip = A_s[w][j][p], aiq = A_s[w][j][q];
            A_s[w][j][p] = c * aip - s * aiq;
            A_s[w][j][q] = s * aip + c * aiq;
            const float uip = U_s[w][j][p], uiq = U_s[w][j][q];
            U_s[w][j][p] = c * uip - s * uiq;
            U_s[w][j][q] = s * uip + c * uiq;
          }
        }
      }
    }
  }
  wave_fence();

  // log of rectified eigenvalues
  if (j < DOUT) l_s[w][j] = __logf(fmaxf(A_s[w][j][j], 1e-4f));
  wave_fence();

  // M = U diag(logS) U^T; upper-triangular (row-major triu order) -> v_s
  if (j < DOUT) {
    for (int i = 0; i <= j; ++i) {
      float acc = 0.0f;
      for (int k = 0; k < DOUT; ++k)
        acc += U_s[w][i][k] * l_s[w][k] * U_s[w][j][k];
      const int idx = i * DOUT - (i * (i - 1)) / 2 + (j - i);
      v_s[w][idx] = acc;
    }
  }
  wave_fence();

  // out[b] = v @ lin_w^T + lin_b
  if (lane < 10) {
    float acc = lb_s[lane];
    const float* lw = &lw_s[lane * 210];
    for (int t = 0; t < 210; ++t) acc += lw[t] * v_s[w][t];
    out[(size_t)b * 10 + lane] = acc;
  }
}

extern "C" void kernel_launch(void* const* d_in, const int* in_sizes, int n_in,
                              void* d_out, int out_size, void* d_ws, size_t ws_size,
                              hipStream_t stream) {
  const float* x     = (const float*)d_in[0];   // (4096, 64, 256) f32
  const float* W     = (const float*)d_in[1];   // (64, 20) f32
  const float* lin_w = (const float*)d_in[2];   // (10, 210) f32
  const float* lin_b = (const float*)d_in[3];   // (10,) f32
  float* Y   = (float*)d_ws;                    // 4096*400 f32 = 6.55 MB scratch
  float* out = (float*)d_out;                   // (4096, 10) f32

  cifarnet_y_kernel<<<4096, 256, 0, stream>>>(x, W, Y);
  cifarnet_eig_kernel<<<512, 256, 0, stream>>>(Y, lin_w, lin_b, out);
}
